// LSTMEncoder_55868934586820
// MI455X (gfx1250) — compile-verified, matrix operand order checked by reading, not run on hardware
//
#include <hip/hip_runtime.h>
#include <hip/hip_bf16.h>
#include <math.h>

typedef __bf16 bf16;
typedef __attribute__((ext_vector_type(16))) __bf16 v16bf;
typedef __attribute__((ext_vector_type(8)))  float  v8f;

#define B_ 256
#define T_ 128
#define D_ 300
#define H_ 512
#define KP_LSTM 832   // D_+H_ = 812 padded to multiple of 32
#define N_LSTM 2048   // 4*H_

// ---------------------------------------------------------------------------
// Convert fp32 weight matrix [K,N] (row-major) into hi/lo bf16 fragments,
// pre-swizzled into the exact per-lane B-matrix layout of
// V_WMMA_F32_16X16X32_BF16:
//   tile(ktile,ntile) of 32(K)x16(N); lane L holds column n=L%16;
//   lanes 0-15 cover K 0-7 (vgpr0-3) and 16-23 (vgpr4-7);
//   lanes 16-31 cover K 8-15 and 24-31.
// Each lane's 16 bf16 are stored contiguously -> fragment load = 2x b128.
// ---------------------------------------------------------------------------
__global__ __launch_bounds__(256) void k_convert_swz(
    const float* __restrict__ W, int K, int N, int Kp,
    bf16* __restrict__ hi, bf16* __restrict__ lo) {
  int e = blockIdx.x * blockDim.x + threadIdx.x;
  int total = Kp * N;
  if (e >= total) return;
  int ntiles = N >> 4;
  int tile = e >> 9;          // 512 bf16 per 32x16 tile
  int slot = e & 511;
  int lane = slot >> 4;
  int idx  = slot & 15;
  int ktile = tile / ntiles;
  int ntile = tile - ktile * ntiles;
  int v = idx >> 1, p = idx & 1;
  int kin = ((v < 4) ? (2 * v + p) : (16 + 2 * (v - 4) + p)) + ((lane >> 4) << 3);
  int k = ktile * 32 + kin;
  int n = (ntile << 4) + (lane & 15);
  float w = (k < K) ? W[(size_t)k * N + n] : 0.0f;
  bf16 h = (bf16)w;
  hi[e] = h;
  lo[e] = (bf16)(w - (float)h);
}

// ---------------------------------------------------------------------------
// WMMA GEMM: out[M,N] = act( A[M,Kp] @ W[Kp,N] + bias[N] )
//   A: fp32 row-major, lda == Kp (pre-padded with zeros where needed)
//   W: pre-swizzled hi/lo bf16 fragments (see k_convert_swz)
//   block = 128 threads (4 waves). Each wave owns FOUR 16x16 N-tiles
//   (amortizes the A-fragment hi/lo split over 12 WMMAs per K-chunk);
//   block tile = 16 (M) x 256 (N).
//   fp32 accumulation with bf16 split products: Ah*Bh + Ah*Bl + Al*Bh
// ---------------------------------------------------------------------------
__global__ __launch_bounds__(128) void k_gemm(
    const float* __restrict__ A, int lda,
    const bf16* __restrict__ Whi, const bf16* __restrict__ Wlo,
    const float* __restrict__ bias,
    float* __restrict__ out, int ldo, int coloff,
    int Kp, int N, int act) {
  __shared__ float As[16][33];   // 16 rows x 32 K, padded stride
  const int tid  = threadIdx.x;
  const int lane = tid & 31;
  const int wave = tid >> 5;
  const int ntile0 = (blockIdx.x << 4) + (wave << 2);  // 4 consecutive n-tiles
  const int mtile = blockIdx.y;
  const int ntiles = N >> 4;
  const float* Arow = A + (size_t)(mtile << 4) * lda;
  const int mm    = lane & 15;
  const int khalf = (lane >> 4) << 3;

  v8f acc0 = {}, acc1 = {}, acc2 = {}, acc3 = {};
  const int nk = Kp >> 5;
  for (int kc = 0; kc < nk; ++kc) {
    // cooperative stage of the 16x32 fp32 A tile (shared by all 4 waves)
#pragma unroll
    for (int i = 0; i < 4; ++i) {
      int li = tid + (i << 7);
      int r = li >> 5, kk = li & 31;
      As[r][kk] = Arow[(size_t)r * lda + (kc << 5) + kk];
    }
    __syncthreads();

    // build A fragments (documented 16-bit 16x32 A layout), split hi/lo
    v16bf a_hi, a_lo;
#pragma unroll
    for (int i = 0; i < 16; ++i) {
      int v = i >> 1, p = i & 1;
      int kin = ((v < 4) ? (2 * v + p) : (16 + 2 * (v - 4) + p)) + khalf;
      float x = As[mm][kin];
      bf16 h = (bf16)x;
      a_hi[i] = h;
      a_lo[i] = (bf16)(x - (float)h);
    }

    // fragment loads: 4 consecutive n-tiles, contiguous 32B per lane each
    size_t fbase = ((size_t)(kc * ntiles + ntile0) << 9) + ((size_t)lane << 4);
    v16bf bh0 = *(const v16bf*)(Whi + fbase);
    v16bf bh1 = *(const v16bf*)(Whi + fbase + 512);
    v16bf bh2 = *(const v16bf*)(Whi + fbase + 1024);
    v16bf bh3 = *(const v16bf*)(Whi + fbase + 1536);
    v16bf bl0 = *(const v16bf*)(Wlo + fbase);
    v16bf bl1 = *(const v16bf*)(Wlo + fbase + 512);
    v16bf bl2 = *(const v16bf*)(Wlo + fbase + 1024);
    v16bf bl3 = *(const v16bf*)(Wlo + fbase + 1536);

    acc0 = __builtin_amdgcn_wmma_f32_16x16x32_bf16(false, a_hi, false, bh0, (short)0, acc0, false, false);
    acc1 = __builtin_amdgcn_wmma_f32_16x16x32_bf16(false, a_hi, false, bh1, (short)0, acc1, false, false);
    acc2 = __builtin_amdgcn_wmma_f32_16x16x32_bf16(false, a_hi, false, bh2, (short)0, acc2, false, false);
    acc3 = __builtin_amdgcn_wmma_f32_16x16x32_bf16(false, a_hi, false, bh3, (short)0, acc3, false, false);
    acc0 = __builtin_amdgcn_wmma_f32_16x16x32_bf16(false, a_hi, false, bl0, (short)0, acc0, false, false);
    acc1 = __builtin_amdgcn_wmma_f32_16x16x32_bf16(false, a_hi, false, bl1, (short)0, acc1, false, false);
    acc2 = __builtin_amdgcn_wmma_f32_16x16x32_bf16(false, a_hi, false, bl2, (short)0, acc2, false, false);
    acc3 = __builtin_amdgcn_wmma_f32_16x16x32_bf16(false, a_hi, false, bl3, (short)0, acc3, false, false);
    acc0 = __builtin_amdgcn_wmma_f32_16x16x32_bf16(false, a_lo, false, bh0, (short)0, acc0, false, false);
    acc1 = __builtin_amdgcn_wmma_f32_16x16x32_bf16(false, a_lo, false, bh1, (short)0, acc1, false, false);
    acc2 = __builtin_amdgcn_wmma_f32_16x16x32_bf16(false, a_lo, false, bh2, (short)0, acc2, false, false);
    acc3 = __builtin_amdgcn_wmma_f32_16x16x32_bf16(false, a_lo, false, bh3, (short)0, acc3, false, false);
    __syncthreads();
  }

  // epilogue: C/D layout — vgpr r: lanes0-15 M=r, lanes16-31 M=8+r; N=lane%16
  int rbase = (mtile << 4) + ((lane >> 4) << 3);
  v8f accs[4] = {acc0, acc1, acc2, acc3};
#pragma unroll
  for (int j = 0; j < 4; ++j) {
    int col = ((ntile0 + j) << 4) + (lane & 15);
    float bv = bias ? bias[col] : 0.0f;
#pragma unroll
    for (int r = 0; r < 8; ++r) {
      float x = accs[j][r] + bv;
      if (act == 1) x = tanhf(x);
      out[(size_t)(rbase + r) * ldo + coloff + col] = x;
    }
  }
}

// ---------------------------------------------------------------------------
// Per-step input pack: A = [x_t (D), h (H), zero-pad to 832] for both LSTMs
// ---------------------------------------------------------------------------
__global__ __launch_bounds__(256) void k_pack(
    const float* __restrict__ prem, const float* __restrict__ hyp,
    const float* __restrict__ h_p, const float* __restrict__ h_h,
    float* __restrict__ A_p, float* __restrict__ A_h, int t) {
  int idx = blockIdx.x * blockDim.x + threadIdx.x;
  const int total = B_ * KP_LSTM;
  if (idx >= 2 * total) return;
  int which = (idx >= total);
  int e = which ? idx - total : idx;
  int b = e / KP_LSTM, k = e - b * KP_LSTM;
  const float* x = which ? hyp : prem;
  const float* h = which ? h_h : h_p;
  float v;
  if (k < D_)            v = x[((size_t)b * T_ + t) * D_ + k];
  else if (k < D_ + H_)  v = h[(size_t)b * H_ + (k - D_)];
  else                   v = 0.0f;
  (which ? A_h : A_p)[e] = v;
}

// ---------------------------------------------------------------------------
// LSTM gate nonlinearity (TF BasicLSTMCell: gates i,j,f,o, forget_bias=1.0)
// blockIdx.z selects premise (0) / hypothesis (1) LSTM
// ---------------------------------------------------------------------------
__global__ __launch_bounds__(256) void k_gate(
    const float* __restrict__ z_p, const float* __restrict__ z_h,
    float* __restrict__ h_p, float* __restrict__ c_p,
    float* __restrict__ h_h, float* __restrict__ c_h) {
  int idx = blockIdx.x * blockDim.x + threadIdx.x;
  if (idx >= B_ * H_) return;
  int b = idx >> 9, u = idx & 511;
  const float* z = blockIdx.z ? z_h : z_p;
  float* h = blockIdx.z ? h_h : h_p;
  float* c = blockIdx.z ? c_h : c_p;
  size_t zb = (size_t)b * N_LSTM;
  float zi = z[zb + u];
  float zj = z[zb + 512 + u];
  float zf = z[zb + 1024 + u];
  float zo = z[zb + 1536 + u];
  float sf = 1.0f / (1.0f + __expf(-(zf + 1.0f)));
  float si = 1.0f / (1.0f + __expf(-zi));
  float so = 1.0f / (1.0f + __expf(-zo));
  float cv = c[idx] * sf + si * tanhf(zj);
  c[idx] = cv;
  h[idx] = tanhf(cv) * so;
}

// ---------------------------------------------------------------------------
// Final projection [256,1024] @ [1024,3] + b  (tiny; VALU dot products)
// ---------------------------------------------------------------------------
__global__ __launch_bounds__(256) void k_out(
    const float* __restrict__ X, const float* __restrict__ W,
    const float* __restrict__ bo, float* __restrict__ out) {
  int idx = blockIdx.x * blockDim.x + threadIdx.x;
  if (idx >= B_ * 3) return;
  int b = idx / 3, l = idx - b * 3;
  float s = bo[l];
  const float* x = X + (size_t)b * 1024;
  for (int k = 0; k < 1024; ++k) s += x[k] * W[k * 3 + l];
  out[idx] = s;
}

// ---------------------------------------------------------------------------
extern "C" void kernel_launch(void* const* d_in, const int* in_sizes, int n_in,
                              void* d_out, int out_size, void* d_ws, size_t ws_size,
                              hipStream_t stream) {
  (void)in_sizes; (void)n_in; (void)out_size; (void)ws_size;
  const float* premises = (const float*)d_in[0];
  const float* hypoths  = (const float*)d_in[1];
  const float* W_lstm_p = (const float*)d_in[2];
  const float* b_lstm_p = (const float*)d_in[3];
  const float* W_lstm_h = (const float*)d_in[4];
  const float* b_lstm_h = (const float*)d_in[5];
  const float* W_red_p  = (const float*)d_in[6];
  const float* b_red_p  = (const float*)d_in[7];
  const float* W_red_h  = (const float*)d_in[8];
  const float* b_red_h  = (const float*)d_in[9];
  const float* W1 = (const float*)d_in[10];
  const float* b1 = (const float*)d_in[11];
  const float* W2 = (const float*)d_in[12];
  const float* b2 = (const float*)d_in[13];
  const float* W3 = (const float*)d_in[14];
  const float* b3 = (const float*)d_in[15];
  const float* W_out = (const float*)d_in[16];
  const float* b_out = (const float*)d_in[17];
  float* out = (float*)d_out;

  // ---- workspace carve-out ----
  char* ws = (char*)d_ws;
  size_t off = 0;
  auto carve = [&](size_t bytes) -> char* {
    char* p = ws + off;
    off = (off + bytes + 255) & ~(size_t)255;
    return p;
  };
  const size_t SZ_WLSTM = (size_t)KP_LSTM * N_LSTM * sizeof(bf16);
  const size_t SZ_WRED  = (size_t)512 * 512 * sizeof(bf16);
  const size_t SZ_WMLP  = (size_t)1024 * 1024 * sizeof(bf16);

  bf16* wp_hi = (bf16*)carve(SZ_WLSTM);  bf16* wp_lo = (bf16*)carve(SZ_WLSTM);
  bf16* wh_hi = (bf16*)carve(SZ_WLSTM);  bf16* wh_lo = (bf16*)carve(SZ_WLSTM);
  bf16* wrp_hi = (bf16*)carve(SZ_WRED);  bf16* wrp_lo = (bf16*)carve(SZ_WRED);
  bf16* wrh_hi = (bf16*)carve(SZ_WRED);  bf16* wrh_lo = (bf16*)carve(SZ_WRED);
  bf16* w1_hi = (bf16*)carve(SZ_WMLP);   bf16* w1_lo = (bf16*)carve(SZ_WMLP);
  bf16* w2_hi = (bf16*)carve(SZ_WMLP);   bf16* w2_lo = (bf16*)carve(SZ_WMLP);
  bf16* w3_hi = (bf16*)carve(SZ_WMLP);   bf16* w3_lo = (bf16*)carve(SZ_WMLP);
  float* A_p = (float*)carve((size_t)B_ * KP_LSTM * sizeof(float));
  float* A_h = (float*)carve((size_t)B_ * KP_LSTM * sizeof(float));
  float* z_p = (float*)carve((size_t)B_ * N_LSTM * sizeof(float));
  float* z_h = (float*)carve((size_t)B_ * N_LSTM * sizeof(float));
  float* states = (float*)carve((size_t)4 * B_ * H_ * sizeof(float));
  float* h_p = states;
  float* c_p = states + (size_t)B_ * H_;
  float* h_h = states + (size_t)2 * B_ * H_;
  float* c_h = states + (size_t)3 * B_ * H_;
  float* xcat = (float*)carve((size_t)B_ * 1024 * sizeof(float));
  float* tA   = (float*)carve((size_t)B_ * 1024 * sizeof(float));
  float* tB   = (float*)carve((size_t)B_ * 1024 * sizeof(float));

  // ---- weight conversion to swizzled hi/lo bf16 (every call; deterministic) ----
  {
    int n;
    n = KP_LSTM * N_LSTM;
    k_convert_swz<<<(n + 255) / 256, 256, 0, stream>>>(W_lstm_p, D_ + H_, N_LSTM, KP_LSTM, wp_hi, wp_lo);
    k_convert_swz<<<(n + 255) / 256, 256, 0, stream>>>(W_lstm_h, D_ + H_, N_LSTM, KP_LSTM, wh_hi, wh_lo);
    n = 512 * 512;
    k_convert_swz<<<(n + 255) / 256, 256, 0, stream>>>(W_red_p, 512, 512, 512, wrp_hi, wrp_lo);
    k_convert_swz<<<(n + 255) / 256, 256, 0, stream>>>(W_red_h, 512, 512, 512, wrh_hi, wrh_lo);
    n = 1024 * 1024;
    k_convert_swz<<<(n + 255) / 256, 256, 0, stream>>>(W1, 1024, 1024, 1024, w1_hi, w1_lo);
    k_convert_swz<<<(n + 255) / 256, 256, 0, stream>>>(W2, 1024, 1024, 1024, w2_hi, w2_lo);
    k_convert_swz<<<(n + 255) / 256, 256, 0, stream>>>(W3, 1024, 1024, 1024, w3_hi, w3_lo);
  }

  // zero h/c states
  hipMemsetAsync(states, 0, (size_t)4 * B_ * H_ * sizeof(float), stream);

  // ---- recurrent loop: pack -> gemm(z_p) -> gemm(z_h) -> gates ----
  const dim3 gemm_grid_lstm(N_LSTM / 256, B_ / 16);   // 8 x 16 blocks
  const int pack_blocks = (2 * B_ * KP_LSTM + 255) / 256;
  const dim3 gate_grid((B_ * H_ + 255) / 256, 1, 2);
  for (int t = 0; t < T_; ++t) {
    k_pack<<<pack_blocks, 256, 0, stream>>>(premises, hypoths, h_p, h_h, A_p, A_h, t);
    k_gemm<<<gemm_grid_lstm, 128, 0, stream>>>(A_p, KP_LSTM, wp_hi, wp_lo, b_lstm_p,
                                               z_p, N_LSTM, 0, KP_LSTM, N_LSTM, 0);
    k_gemm<<<gemm_grid_lstm, 128, 0, stream>>>(A_h, KP_LSTM, wh_hi, wh_lo, b_lstm_h,
                                               z_h, N_LSTM, 0, KP_LSTM, N_LSTM, 0);
    k_gate<<<gate_grid, 256, 0, stream>>>(z_p, z_h, h_p, c_p, h_h, c_h);
  }

  // ---- head ----
  const dim3 grid_red(512 / 256, B_ / 16);
  k_gemm<<<grid_red, 128, 0, stream>>>(h_p, 512, wrp_hi, wrp_lo, b_red_p,
                                       xcat, 1024, 0, 512, 512, 1);
  k_gemm<<<grid_red, 128, 0, stream>>>(h_h, 512, wrh_hi, wrh_lo, b_red_h,
                                       xcat, 1024, 512, 512, 512, 1);
  const dim3 grid_mlp(1024 / 256, B_ / 16);
  k_gemm<<<grid_mlp, 128, 0, stream>>>(xcat, 1024, w1_hi, w1_lo, b1, tA, 1024, 0, 1024, 1024, 1);
  k_gemm<<<grid_mlp, 128, 0, stream>>>(tA,   1024, w2_hi, w2_lo, b2, tB, 1024, 0, 1024, 1024, 1);
  k_gemm<<<grid_mlp, 128, 0, stream>>>(tB,   1024, w3_hi, w3_lo, b3, tA, 1024, 0, 1024, 1024, 1);
  k_out<<<(B_ * 3 + 255) / 256, 256, 0, stream>>>(tA, W_out, b_out, out);
}
